// MoTLlamaDecoderLayer_6219112644804
// MI455X (gfx1250) — matrix operand order repaired and checked by reading, not compiled, not run on hardware
//
#include <hip/hip_runtime.h>

// ---------------------------------------------------------------------------
// MoT Llama decoder layer for MI455X (gfx1250, wave32, WMMA bf16).
// B=2 S=1024 D=2048 H=32 HD=64 KVH=8 I=8192.
// GEMMs: v_wmma_f32_16x16x32_bf16, fp32 accumulate, bf16 operands,
// async global->LDS double-buffered A tiles, transposed-LDS B tiles so all
// fragment gathers are ds_load_b128.
// ---------------------------------------------------------------------------

#define BB 2
#define SS 1024
#define DD 2048
#define HH 32
#define HD 64
#define KVH 8
#define II 8192
#define ROWS (BB * SS)          // 2048 token rows

typedef __attribute__((ext_vector_type(16))) __bf16 v16bf;
typedef __attribute__((ext_vector_type(8)))  float  v8f;
typedef __attribute__((ext_vector_type(4)))  int    v4i;

// --- async global->LDS (gfx1250) with guarded fallback ----------------------
#if defined(__has_builtin)
#  if __has_builtin(__builtin_amdgcn_global_load_async_to_lds_b128) && \
      __has_builtin(__builtin_amdgcn_s_wait_asynccnt)
#    define USE_ASYNC_LDS 1
#  endif
#endif
#ifndef USE_ASYNC_LDS
#  define USE_ASYNC_LDS 0
#endif

#if USE_ASYNC_LDS
typedef __attribute__((address_space(1))) v4i GV4;   // global int4
typedef __attribute__((address_space(3))) v4i LV4;   // LDS int4
#endif

__device__ __forceinline__ void copy16_g2l(const unsigned short* g, unsigned short* l) {
#if USE_ASYNC_LDS
    __builtin_amdgcn_global_load_async_to_lds_b128(
        (GV4*)(unsigned long long)g,
        (LV4*)(unsigned int)(unsigned long long)l, 0, 0);
#else
    uint4 d = *(const uint4*)g;
    unsigned int* sp = (unsigned int*)l;
    sp[0] = d.x; sp[1] = d.y; sp[2] = d.z; sp[3] = d.w;
#endif
}

__device__ __forceinline__ void wait_async_zero() {
#if USE_ASYNC_LDS
    __builtin_amdgcn_s_wait_asynccnt(0);
#endif
}

__device__ __forceinline__ unsigned short f2bf(float f) {
    union { float f; unsigned int u; } v; v.f = f;
    unsigned int r = v.u + 0x7FFFu + ((v.u >> 16) & 1u);   // round-nearest-even
    return (unsigned short)(r >> 16);
}

__device__ __forceinline__ v16bf pack_bf16(const unsigned int* u) {
    union { unsigned int u[8]; v16bf v; } x;
#pragma unroll
    for (int i = 0; i < 8; ++i) x.u[i] = u[i];
    return x.v;
}

__device__ __forceinline__ v16bf pack_bf16_q(uint4 a, uint4 b) {
    union { uint4 q[2]; v16bf v; } x;
    x.q[0] = a; x.q[1] = b;
    return x.v;
}

__device__ __forceinline__ v8f v8f_zero() {
    v8f z;
#pragma unroll
    for (int i = 0; i < 8; ++i) z[i] = 0.0f;
    return z;
}

// ---------------------------------------------------------------------------
// RMSNorm with optional row mask (maskA && !maskB), fp32 in -> bf16 out.
// ---------------------------------------------------------------------------
__global__ __launch_bounds__(256) void rmsnorm_mask_kernel(
    const float* __restrict__ x, const float* __restrict__ w,
    const unsigned char* __restrict__ maskA, const unsigned char* __restrict__ maskB,
    unsigned short* __restrict__ out, int D)
{
    int row = blockIdx.x;
    const float* xr = x + (size_t)row * D;
    __shared__ float red[8];
    __shared__ float stot;
    float ss = 0.0f;
    for (int i = threadIdx.x; i < D; i += 256) { float v = xr[i]; ss += v * v; }
#pragma unroll
    for (int off = 16; off >= 1; off >>= 1) ss += __shfl_xor(ss, off, 32);
    if ((threadIdx.x & 31) == 0) red[threadIdx.x >> 5] = ss;
    __syncthreads();
    if (threadIdx.x == 0) {
        float t = 0.0f;
        for (int i = 0; i < 8; ++i) t += red[i];
        stot = t;
    }
    __syncthreads();
    float scale = rsqrtf(stot / (float)D + 1e-6f);
    bool act = true;
    if (maskA) {
        act = maskA[row] != 0;
        if (maskB) act = act && (maskB[row] == 0);
    }
    if (!act) scale = 0.0f;
    unsigned short* orow = out + (size_t)row * D;
    for (int i = threadIdx.x; i < D; i += 256) orow[i] = f2bf(xr[i] * scale * w[i]);
}

// ---------------------------------------------------------------------------
// WMMA bf16 GEMM: C[M,N] = A[M,K](bf16) * Bw[K,N](f32 -> bf16 on the fly),
// fused epilogues. 128x128x32 block tile, 8 waves (2x4), 64x32 per wave.
// A tiles: async global->LDS double buffer (row stride 40 halves).
// B tiles: register staged, converted, stored TRANSPOSED (Bt[n][k]) so the
// B fragment of each lane is two 16-byte runs of its Bt row -> 2x ds_load_b128.
// ---------------------------------------------------------------------------
#define GM_BM 128
#define GM_BN 128
#define GM_BK 32
#define AS_STR 40   // halves: 80B rows, 16B-aligned, conflict-free b128 gathers
#define BT_STR 40   // halves: same property for transposed B rows

#define EP_STORE_F32   0
#define EP_ACCUM_F32   1
#define EP_ACCUM_BF16  2
#define EP_SILU_BF16   3
#define EP_RESID_MASK  4
#define EP_RESID       5

__global__ __launch_bounds__(256) void gemm_bf16_kernel(
    const unsigned short* __restrict__ A, const float* __restrict__ Bw,
    int M, int N, int K, int mode,
    float* __restrict__ outF, unsigned short* __restrict__ outH,
    const float* __restrict__ Cin, const float* __restrict__ Gin,
    const float* __restrict__ resid, const unsigned char* __restrict__ mask)
{
    __shared__ unsigned short As[2][GM_BM * AS_STR];   // 2 x 10240 B
    __shared__ unsigned short Bt[2][GM_BN * BT_STR];   // 2 x 10240 B (transposed)

    const int tid  = threadIdx.x;
    const int wave = tid >> 5;
    const int l    = tid & 31;
    const int lane = l & 15;
    const int half = l >> 4;
    const int wm   = wave >> 2;   // 0..1  (64-row slabs)
    const int wn   = wave & 3;    // 0..3  (32-col slabs)
    const int r0   = blockIdx.y * GM_BM;
    const int c0   = blockIdx.x * GM_BN;

    // copy geometry
    int arow[2], acol[2];
#pragma unroll
    for (int p = 0; p < 2; ++p) { int lin = tid + p * 256; arow[p] = lin >> 2; acol[p] = (lin & 3) << 3; }
    int brow[4], bcol[4];
#pragma unroll
    for (int p = 0; p < 4; ++p) { int lin = tid + p * 256; brow[p] = lin >> 5; bcol[p] = (lin & 31) << 2; }

    v8f acc[4][2];
#pragma unroll
    for (int mt = 0; mt < 4; ++mt)
#pragma unroll
        for (int nt = 0; nt < 2; ++nt) acc[mt][nt] = v8f_zero();

    const int nk = K / GM_BK;
    float4 breg[4];

    // prologue: stage tile 0
#pragma unroll
    for (int p = 0; p < 2; ++p)
        copy16_g2l(A + (size_t)(r0 + arow[p]) * K + acol[p],
                   &As[0][arow[p] * AS_STR + acol[p]]);
#pragma unroll
    for (int p = 0; p < 4; ++p)
        breg[p] = *(const float4*)(Bw + (size_t)brow[p] * N + c0 + bcol[p]);

    for (int i = 0; i < nk; ++i) {
        const int cur = i & 1;
        const int kn  = (i + 1) * GM_BK;

        // convert + transpose-store staged B regs for tile i: Bt[n][k]
#pragma unroll
        for (int p = 0; p < 4; ++p) {
            Bt[cur][(bcol[p] + 0) * BT_STR + brow[p]] = f2bf(breg[p].x);
            Bt[cur][(bcol[p] + 1) * BT_STR + brow[p]] = f2bf(breg[p].y);
            Bt[cur][(bcol[p] + 2) * BT_STR + brow[p]] = f2bf(breg[p].z);
            Bt[cur][(bcol[p] + 3) * BT_STR + brow[p]] = f2bf(breg[p].w);
        }
        // start B global loads for tile i+1 (latency hidden under compute)
        if (i + 1 < nk) {
#pragma unroll
            for (int p = 0; p < 4; ++p)
                breg[p] = *(const float4*)(Bw + (size_t)(kn + brow[p]) * N + c0 + bcol[p]);
        }
        if (i + 2 < nk)
            __builtin_prefetch(Bw + (size_t)(kn + GM_BK + brow[0]) * N + c0 + bcol[0], 0, 3);

        wait_async_zero();     // this wave's async A copies for tile i landed
        __syncthreads();       // whole tile resident; previous buffer free

        // kick async A copies for tile i+1 into the other buffer
        if (i + 1 < nk) {
#pragma unroll
            for (int p = 0; p < 2; ++p)
                copy16_g2l(A + (size_t)(r0 + arow[p]) * K + kn + acol[p],
                           &As[1 - cur][arow[p] * AS_STR + acol[p]]);
        }

        // --- fragments: all gathers are 16B LDS loads ---
        v16bf af[4];
#pragma unroll
        for (int mt = 0; mt < 4; ++mt) {
            int m = wm * 64 + mt * 16 + lane;
            const unsigned short* ap = &As[cur][m * AS_STR + (half << 3)];
            uint4 q0 = *(const uint4*)(ap);        // k = half*8 + 0..7
            uint4 q1 = *(const uint4*)(ap + 16);   // k = 16 + half*8 + 0..7
            af[mt] = pack_bf16_q(q0, q1);
        }
        v16bf bfr[2];
#pragma unroll
        for (int nt = 0; nt < 2; ++nt) {
            int n = wn * 32 + nt * 16 + lane;
            const unsigned short* bp = &Bt[cur][n * BT_STR + (half << 3)];
            uint4 q0 = *(const uint4*)(bp);        // k = half*8 + 0..7
            uint4 q1 = *(const uint4*)(bp + 16);   // k = 16 + half*8 + 0..7
            bfr[nt] = pack_bf16_q(q0, q1);
        }
#pragma unroll
        for (int mt = 0; mt < 4; ++mt)
#pragma unroll
            for (int nt = 0; nt < 2; ++nt)
                acc[mt][nt] = __builtin_amdgcn_wmma_f32_16x16x32_bf16(
                    false, af[mt], false, bfr[nt], (short)0, acc[mt][nt], false, false);
    }

    // --- epilogue (C/D layout: lane = N, VGPR r = M row within half) ---
#pragma unroll
    for (int mt = 0; mt < 4; ++mt) {
#pragma unroll
        for (int nt = 0; nt < 2; ++nt) {
            int col = c0 + wn * 32 + nt * 16 + lane;
#pragma unroll
            for (int r = 0; r < 8; ++r) {
                int row = r0 + wm * 64 + mt * 16 + half * 8 + r;
                size_t idx = (size_t)row * N + col;
                float v = acc[mt][nt][r];
                switch (mode) {
                    case EP_STORE_F32:  outF[idx] = v; break;
                    case EP_ACCUM_F32:  outF[idx] = v + Cin[idx]; break;
                    case EP_ACCUM_BF16: outH[idx] = f2bf(v + Cin[idx]); break;
                    case EP_SILU_BF16: {
                        float g = Gin[idx];
                        float s = g / (1.0f + __expf(-g));
                        outH[idx] = f2bf(s * v);
                    } break;
                    case EP_RESID_MASK: outF[idx] = resid[idx] + (mask[row] ? v : 0.0f); break;
                    default:            outF[idx] = resid[idx] + v; break;
                }
            }
        }
    }
}

// ---------------------------------------------------------------------------
// RoPE: f32 [B,S,nh,HD] -> bf16, pairing (d, d+32) with theta=10000.
// ---------------------------------------------------------------------------
__global__ __launch_bounds__(256) void rope_kernel(
    const float* __restrict__ src, unsigned short* __restrict__ dst, int nh, int total)
{
    int gid = blockIdx.x * 256 + threadIdx.x;
    if (gid >= total) return;
    int f = gid & 31;
    int t = gid >> 5;
    int hh = t % nh; t /= nh;
    int s = t % SS;
    int b = t / SS;
    size_t base = (((size_t)(b * SS + s)) * nh + hh) * HD;
    float x1 = src[base + f], x2 = src[base + 32 + f];
    float inv = __powf(10000.0f, -((float)(2 * f)) / 64.0f);
    float ang = (float)s * inv;
    float c = __cosf(ang), si = __sinf(ang);
    dst[base + f]      = f2bf(x1 * c - x2 * si);
    dst[base + 32 + f] = f2bf(x2 * c + x1 * si);
}

// ---------------------------------------------------------------------------
// Flash attention, one wave per (b, h, 16-query tile). Causal, GQA (h/4).
// QK^T and PV via v_wmma_f32_16x16x32_bf16; P re-layout via LDS.
// ---------------------------------------------------------------------------
__global__ __launch_bounds__(32) void attn_kernel(
    const unsigned short* __restrict__ qb, const unsigned short* __restrict__ kb,
    const unsigned short* __restrict__ vb, unsigned short* __restrict__ ob)
{
    const int bid = blockIdx.x;
    const int qt  = bid & 63;          // S/16 = 64 query tiles
    const int h   = (bid >> 6) & 31;
    const int b   = bid >> 11;
    const int l   = threadIdx.x;
    const int lane = l & 15;
    const int half = l >> 4;
    const int kvh  = h >> 2;           // H/KVH = 4

    __shared__ unsigned short Ps[16 * 34];

    v16bf qf[2];
    {
        int qrow = qt * 16 + lane;
        const unsigned short* qp = qb + (((size_t)(b * SS + qrow)) * HH + h) * HD;
#pragma unroll
        for (int c = 0; c < 2; ++c) {
            unsigned int u[8];
#pragma unroll
            for (int v = 0; v < 8; ++v) {
                int k = ((v >> 2) << 4) + (half << 3) + ((v & 3) << 1);
                u[v] = *(const unsigned int*)(qp + c * 32 + k);
            }
            qf[c] = pack_bf16(u);
        }
    }

    v8f o[4];
#pragma unroll
    for (int c = 0; c < 4; ++c) o[c] = v8f_zero();
    float mx[8], sm[8];
#pragma unroll
    for (int r = 0; r < 8; ++r) { mx[r] = -1e30f; sm[r] = 0.0f; }

    const int nkb = (qt + 2) >> 1;
    for (int jb = 0; jb < nkb; ++jb) {
        const int k0 = jb * 32;
        v8f s[2];
#pragma unroll
        for (int t = 0; t < 2; ++t) {
            int key = k0 + t * 16 + lane;
            const unsigned short* kp = kb + (((size_t)(b * SS + key)) * KVH + kvh) * HD;
            s[t] = v8f_zero();
#pragma unroll
            for (int c = 0; c < 2; ++c) {
                const unsigned short* kc = kp + c * 32 + (half << 3);
                uint4 q0 = *(const uint4*)(kc);
                uint4 q1 = *(const uint4*)(kc + 16);
                v16bf kf = pack_bf16_q(q0, q1);
                s[t] = __builtin_amdgcn_wmma_f32_16x16x32_bf16(
                    false, qf[c], false, kf, (short)0, s[t], false, false);
            }
        }
        float corr[8];
#pragma unroll
        for (int r = 0; r < 8; ++r) {
            int row = qt * 16 + half * 8 + r;
            float s0 = s[0][r] * 0.125f; if (k0 + lane > row)      s0 = -1e30f;
            float s1 = s[1][r] * 0.125f; if (k0 + 16 + lane > row) s1 = -1e30f;
            float rm = fmaxf(s0, s1);
#pragma unroll
            for (int off = 1; off < 16; off <<= 1) rm = fmaxf(rm, __shfl_xor(rm, off, 32));
            float nm = fmaxf(mx[r], rm);
            float cr = __expf(mx[r] - nm);
            mx[r] = nm; corr[r] = cr;
            float p0 = __expf(s0 - nm), p1 = __expf(s1 - nm);
            s[0][r] = p0; s[1][r] = p1;
            float ps = p0 + p1;
#pragma unroll
            for (int off = 1; off < 16; off <<= 1) ps += __shfl_xor(ps, off, 32);
            sm[r] = sm[r] * cr + ps;
        }
#pragma unroll
        for (int c = 0; c < 4; ++c)
#pragma unroll
            for (int r = 0; r < 8; ++r) o[c][r] *= corr[r];

#pragma unroll
        for (int r = 0; r < 8; ++r) {
            Ps[(half * 8 + r) * 34 + lane]      = f2bf(s[0][r]);
            Ps[(half * 8 + r) * 34 + 16 + lane] = f2bf(s[1][r]);
        }
        asm volatile("s_wait_dscnt 0" ::: "memory");
        v16bf pf;
        {
            unsigned int u[8];
#pragma unroll
            for (int v = 0; v < 8; ++v) {
                int k = ((v >> 2) << 4) + (half << 3) + ((v & 3) << 1);
                u[v] = *(const unsigned int*)(Ps + lane * 34 + k);
            }
            pf = pack_bf16(u);
        }
#pragma unroll
        for (int c = 0; c < 4; ++c) {
            int d = c * 16 + lane;
            unsigned int u[8];
#pragma unroll
            for (int v = 0; v < 8; ++v) {
                int key0 = k0 + ((v >> 2) << 4) + (half << 3) + ((v & 3) << 1);
                unsigned int lo = vb[(((size_t)(b * SS + key0)) * KVH + kvh) * HD + d];
                unsigned int hi = vb[(((size_t)(b * SS + key0 + 1)) * KVH + kvh) * HD + d];
                u[v] = lo | (hi << 16);
            }
            v16bf vf = pack_bf16(u);
            o[c] = __builtin_amdgcn_wmma_f32_16x16x32_bf16(
                false, pf, false, vf, (short)0, o[c], false, false);
        }
    }
#pragma unroll
    for (int c = 0; c < 4; ++c) {
#pragma unroll
        for (int r = 0; r < 8; ++r) {
            int row = qt * 16 + half * 8 + r;
            int d = c * 16 + lane;
            ob[(((size_t)(b * SS + row)) * HH + h) * HD + d] = f2bf(o[c][r] / sm[r]);
        }
    }
}

// ---------------------------------------------------------------------------
// Host orchestration
// ---------------------------------------------------------------------------
extern "C" void kernel_launch(void* const* d_in, const int* in_sizes, int n_in,
                              void* d_out, int out_size, void* d_ws, size_t ws_size,
                              hipStream_t stream)
{
    (void)in_sizes; (void)n_in; (void)out_size; (void)ws_size;
    const float* hidden0 = (const float*)d_in[0];
    const float* hidden1 = (const float*)d_in[1];
    const unsigned char* m0 = (const unsigned char*)d_in[2];  // jnp bool -> 1 byte
    const unsigned char* m1 = (const unsigned char*)d_in[3];
    const float* wq0 = (const float*)d_in[5];
    const float* wk0 = (const float*)d_in[6];
    const float* wv0 = (const float*)d_in[7];
    const float* wo0 = (const float*)d_in[8];
    const float* wq1 = (const float*)d_in[9];
    const float* wk1 = (const float*)d_in[10];
    const float* wv1 = (const float*)d_in[11];
    const float* wo1 = (const float*)d_in[12];
    const float* ln_in0   = (const float*)d_in[13];
    const float* ln_in1   = (const float*)d_in[14];
    const float* ln_post0 = (const float*)d_in[15];
    const float* ln_post1 = (const float*)d_in[16];
    const float* gate0 = (const float*)d_in[17];
    const float* up0   = (const float*)d_in[18];
    const float* down0 = (const float*)d_in[19];
    const float* gate1 = (const float*)d_in[20];
    const float* up1   = (const float*)d_in[21];
    const float* down1 = (const float*)d_in[22];

    float* out0 = (float*)d_out;
    float* out1 = out0 + (size_t)ROWS * DD;

    char* W = (char*)d_ws;
    size_t off = 0;
    auto take = [&](size_t bytes) -> void* {
        void* p = W + off;
        off += (bytes + 255) & ~(size_t)255;
        return p;
    };
    const size_t M2  = (size_t)ROWS * DD;
    const size_t MKV = (size_t)ROWS * KVH * HD;
    const size_t MI  = (size_t)ROWS * II;
    unsigned short* hm0b = (unsigned short*)take(M2 * 2);
    unsigned short* hm1b = (unsigned short*)take(M2 * 2);
    float* qtmp = (float*)take(M2 * 4);
    float* ktmp = (float*)take(MKV * 4);
    float* vtmp = (float*)take(MKV * 4);
    unsigned short* qbv = (unsigned short*)take(M2 * 2);
    unsigned short* kbv = (unsigned short*)take(MKV * 2);
    unsigned short* vbv = (unsigned short*)take(MKV * 2);
    unsigned short* obv = (unsigned short*)take(M2 * 2);
    float* h0r = (float*)take(M2 * 4);
    float* h1r = (float*)take(M2 * 4);
    float* G   = (float*)take(MI * 4);
    unsigned short* actb = (unsigned short*)take(MI * 2);

    auto gemm = [&](const unsigned short* Aq, const float* Bq, int Mq, int Nq, int Kq,
                    int mode, float* oF, unsigned short* oH, const float* Cin,
                    const float* Gin, const float* resid, const unsigned char* msk) {
        dim3 grid(Nq / GM_BN, Mq / GM_BM);
        gemm_bf16_kernel<<<grid, dim3(256), 0, stream>>>(
            Aq, Bq, Mq, Nq, Kq, mode, oF, oH, Cin, Gin, resid, msk);
    };

    rmsnorm_mask_kernel<<<ROWS, 256, 0, stream>>>(hidden0, ln_in0, m0, m1, hm0b, DD);
    rmsnorm_mask_kernel<<<ROWS, 256, 0, stream>>>(hidden1, ln_in1, m1, nullptr, hm1b, DD);

    gemm(hm0b, wq0, ROWS, DD, DD, EP_STORE_F32, qtmp, nullptr, nullptr, nullptr, nullptr, nullptr);
    gemm(hm1b, wq1, ROWS, DD, DD, EP_ACCUM_F32, qtmp, nullptr, qtmp, nullptr, nullptr, nullptr);
    gemm(hm0b, wk0, ROWS, KVH * HD, DD, EP_STORE_F32, ktmp, nullptr, nullptr, nullptr, nullptr, nullptr);
    gemm(hm1b, wk1, ROWS, KVH * HD, DD, EP_ACCUM_F32, ktmp, nullptr, ktmp, nullptr, nullptr, nullptr);
    gemm(hm0b, wv0, ROWS, KVH * HD, DD, EP_STORE_F32, vtmp, nullptr, nullptr, nullptr, nullptr, nullptr);
    gemm(hm1b, wv1, ROWS, KVH * HD, DD, EP_ACCUM_BF16, nullptr, vbv, vtmp, nullptr, nullptr, nullptr);

    {
        int tq = BB * SS * HH * 32;
        rope_kernel<<<(tq + 255) / 256, 256, 0, stream>>>(qtmp, qbv, HH, tq);
        int tk = BB * SS * KVH * 32;
        rope_kernel<<<(tk + 255) / 256, 256, 0, stream>>>(ktmp, kbv, KVH, tk);
    }

    attn_kernel<<<BB * HH * (SS / 16), 32, 0, stream>>>(qbv, kbv, vbv, obv);

    gemm(obv, wo0, ROWS, DD, DD, EP_RESID_MASK, h0r, nullptr, nullptr, nullptr, hidden0, m0);
    gemm(obv, wo1, ROWS, DD, DD, EP_RESID_MASK, h1r, nullptr, nullptr, nullptr, hidden1, m1);

    rmsnorm_mask_kernel<<<ROWS, 256, 0, stream>>>(h0r, ln_post0, nullptr, nullptr, hm0b, DD);
    gemm(hm0b, gate0, ROWS, II, DD, EP_STORE_F32, G, nullptr, nullptr, nullptr, nullptr, nullptr);
    gemm(hm0b, up0,   ROWS, II, DD, EP_SILU_BF16, nullptr, actb, nullptr, G, nullptr, nullptr);
    gemm(actb, down0, ROWS, DD, II, EP_RESID, out0, nullptr, nullptr, nullptr, h0r, nullptr);

    rmsnorm_mask_kernel<<<ROWS, 256, 0, stream>>>(h1r, ln_post1, nullptr, nullptr, hm1b, DD);
    gemm(hm1b, gate1, ROWS, II, DD, EP_STORE_F32, G, nullptr, nullptr, nullptr, nullptr, nullptr);
    gemm(hm1b, up1,   ROWS, II, DD, EP_SILU_BF16, nullptr, actb, nullptr, G, nullptr, nullptr);
    gemm(actb, down1, ROWS, DD, II, EP_RESID, out1, nullptr, nullptr, nullptr, h1r, nullptr);
}
